// KmeansAttention_86354612453691
// MI455X (gfx1250) — matrix operand order repaired and checked by reading, not compiled, not run on hardware
//
#include <hip/hip_runtime.h>
#include <hip/hip_bf16.h>
#include <math.h>
#include <float.h>

// Problem constants (match reference)
#define BB   2
#define TT   8192
#define HH   8
#define DD   64
#define NCC  128
#define MMEM 4
#define WW   64      // T / NC
#define LK   68      // MMEM + WW (valid attention length)
#define LKP  80      // LK padded to 5*16 for WMMA N tiles
#define LVP  96      // K-dim padding for P@V (multiple of 32)
#define CPB  4       // clusters per selection workgroup

typedef __attribute__((ext_vector_type(16))) _Float16 v16h;
typedef __attribute__((ext_vector_type(8)))  float    v8f;

union AF { v16h v; _Float16 e[16]; };
union CF { v8f  v; float    e[8];  };

// Parameter type of __builtin_amdgcn_global_load_async_to_lds_b128 (from hipcc
// diagnostic: generic pointer to int __vector_size__(4*sizeof(int)))
typedef int async_b128_t __attribute__((__vector_size__(4 * sizeof(int))));

#if defined(__HIP_DEVICE_COMPILE__) && __has_builtin(__builtin_amdgcn_global_load_async_to_lds_b128) && __has_builtin(__builtin_amdgcn_s_wait_asynccnt)
#define HAVE_ASYNC_LDS 1
#else
#define HAVE_ASYNC_LDS 0
#endif

// ---- WMMA fragment loaders (wave32, 16x16x32 f16) -------------------------
// A matrix 16x32 (MxK): lane m = lane&15, half = lane>>4;
// VGPR j<4 : K = 8*half + 2j ; VGPR j>=4 : K = 16 + 8*half + 2(j-4)
__device__ __forceinline__ int kbA(int j, int half) {
  return (j < 4) ? (8 * half + 2 * j) : (16 + 8 * half + 2 * (j - 4));
}
// B matrix 32x16 (KxN): lane n = lane&15, half = lane>>4; same K mapping as A
__device__ __forceinline__ int kbB(int j, int half) {
  return (j < 4) ? (8 * half + 2 * j) : (16 + 8 * half + 2 * (j - 4));
}

__device__ __forceinline__ v16h load_a(const _Float16* base, int ld, int row0, int col0, int lane) {
  int m = row0 + (lane & 15);
  int half = (lane >> 4) & 1;
  AF f;
#pragma unroll
  for (int j = 0; j < 8; ++j) {
    int kb = col0 + kbA(j, half);
    f.e[2 * j]     = base[m * ld + kb];
    f.e[2 * j + 1] = base[m * ld + kb + 1];
  }
  return f.v;
}

// B fragment, source stored N-major (row = n, col = k): B[k][n] = src[n][k]
__device__ __forceinline__ v16h load_b_nt(const _Float16* base, int ld, int n0, int k0, int lane) {
  int n = n0 + (lane & 15);
  int half = (lane >> 4) & 1;
  AF f;
#pragma unroll
  for (int j = 0; j < 8; ++j) {
    int kb = k0 + kbB(j, half);
    f.e[2 * j]     = base[n * ld + kb];
    f.e[2 * j + 1] = base[n * ld + kb + 1];
  }
  return f.v;
}

// ---- Kernel A: aux loss = sum over tokens of min_c ||xn - m_c||^2 ---------
__global__ void kmattn_aux_kernel(const float* __restrict__ q, const float* __restrict__ k,
                                  const float* __restrict__ means, float* __restrict__ loss) {
  __shared__ float sM[NCC * DD];
  __shared__ float sM2[NCC];
  __shared__ float sRed[256];
  int tid = threadIdx.x;
  int gid0 = blockIdx.x * 256;
  int bh = gid0 / (2 * TT);           // all 256 tokens of a block share (b,h)
  int h  = bh % HH;
  const float* mh = means + (size_t)h * NCC * DD;
  for (int i = tid; i < NCC * DD; i += 256) sM[i] = mh[i];
  __syncthreads();
  if (tid < NCC) {
    float s = 0.f;
    for (int d0 = 0; d0 < DD; ++d0) { float mv = sM[tid * DD + d0]; s += mv * mv; }
    sM2[tid] = s;
  }
  __syncthreads();
  int tok2 = (gid0 + tid) % (2 * TT);
  const float* row = ((tok2 < TT) ? q : k) + ((size_t)bh * TT + (tok2 % TT)) * DD;
  float x[DD];
  float ss = 0.f;
#pragma unroll
  for (int d0 = 0; d0 < DD; ++d0) { x[d0] = row[d0]; ss += x[d0] * x[d0]; }
  float nrm = sqrtf(ss);
  float inv = 1.0f / (nrm + 1e-6f);
  float xn2 = nrm * inv; xn2 *= xn2;
  float best = FLT_MAX;
  for (int c = 0; c < NCC; ++c) {
    float dot = 0.f;
#pragma unroll
    for (int d0 = 0; d0 < DD; ++d0) dot += x[d0] * sM[c * DD + d0];
    float d2 = xn2 + sM2[c] - 2.0f * dot * inv;
    best = fminf(best, d2);
  }
  best = fmaxf(best, 0.0f);
  sRed[tid] = best;
  __syncthreads();
  for (int s = 128; s > 0; s >>= 1) {
    if (tid < s) sRed[tid] += sRed[tid + s];
    __syncthreads();
  }
  if (tid == 0) atomicAdd(loss, sRed[0]);
}

// ---- Kernel B: per-cluster top-64 nearest tokens, CPB clusters/block ------
__global__ void kmattn_select_kernel(const float* __restrict__ q, const float* __restrict__ k,
                                     const float* __restrict__ means,
                                     int* __restrict__ idxq, int* __restrict__ idxkv) {
  __shared__ float sMean[CPB * DD];
  __shared__ float sM2[CPB];
  __shared__ float sDist[CPB][TT];    // CPB * 32 KB resident distances
  __shared__ float sRv[256];
  __shared__ int   sRi[256];
  const int NG = NCC / CPB;           // cluster groups per (b,h,side)
  int tid = threadIdx.x;
  int bid = blockIdx.x;
  int side = bid / (BB * HH * NG);
  int rem  = bid % (BB * HH * NG);
  int cg = rem % NG;
  int bh = rem / NG;
  int h  = bh % HH;
  const float* src = side ? k : q;
  int* dst = side ? idxkv : idxq;
  sMean[tid] = means[((size_t)h * NCC + cg * CPB) * DD + tid];  // CPB*DD == 256
  __syncthreads();
  if (tid < CPB) {
    float s = 0.f;
    for (int d0 = 0; d0 < DD; ++d0) { float mv = sMean[tid * DD + d0]; s += mv * mv; }
    sM2[tid] = s;
  }
  __syncthreads();
  const float* base = src + (size_t)bh * TT * DD;
  for (int t2 = tid; t2 < TT; t2 += 256) {
    const float4* row = (const float4*)(base + (size_t)t2 * DD);
    __builtin_prefetch(base + (size_t)(t2 + 256) * DD, 0, 0);
    float ss = 0.f;
    float dt[CPB] = {0.f, 0.f, 0.f, 0.f};
#pragma unroll
    for (int d4 = 0; d4 < DD / 4; ++d4) {
      float4 xv = row[d4];
      ss += xv.x * xv.x + xv.y * xv.y + xv.z * xv.z + xv.w * xv.w;
#pragma unroll
      for (int j = 0; j < CPB; ++j) {
        const float* mj = &sMean[j * DD + d4 * 4];
        dt[j] += xv.x * mj[0] + xv.y * mj[1] + xv.z * mj[2] + xv.w * mj[3];
      }
    }
    float nrm = sqrtf(ss);
    float inv = 1.0f / (nrm + 1e-6f);
    float xn2 = nrm * inv; xn2 *= xn2;
#pragma unroll
    for (int j = 0; j < CPB; ++j) sDist[j][t2] = xn2 + sM2[j] - 2.0f * dt[j] * inv;
  }
  __syncthreads();
  for (int j = 0; j < CPB; ++j) {
    for (int it = 0; it < WW; ++it) {
      float bv = FLT_MAX; int bi = TT;
      for (int t2 = tid; t2 < TT; t2 += 256) {
        float vv = sDist[j][t2];
        if (vv < bv) { bv = vv; bi = t2; }
      }
      sRv[tid] = bv; sRi[tid] = bi;
      __syncthreads();
      for (int s = 128; s > 0; s >>= 1) {
        if (tid < s) {
          if (sRv[tid + s] < sRv[tid] ||
              (sRv[tid + s] == sRv[tid] && sRi[tid + s] < sRi[tid])) {
            sRv[tid] = sRv[tid + s]; sRi[tid] = sRi[tid + s];
          }
        }
        __syncthreads();
      }
      if (tid == 0) {
        int wsel = sRi[0];
        dst[((size_t)bh * NCC + cg * CPB + j) * WW + it] = wsel;
        sDist[j][wsel] = FLT_MAX;
      }
      __syncthreads();
    }
  }
}

// ---- Kernel C: per-cluster attention via v_wmma_f32_16x16x32_f16 ----------
// LDS map (64 KB total, staging overlaid by scores/probs):
//   [0      .. 17408)  Kstage f32 (68x64)   | overlaid later by Ssm f32 (64x80 = 20480 B)
//   [17408  .. 34816)  Vstage f32 (68x64)   | overlaid later by Ph  f16 (64x96 @ 20480..32768)
//   [34816  .. 43008)  Qh  f16 (64x64)
//   [43008  .. 53248)  Kh  f16 (80x64, rows >=68 zero)
//   [53248  .. 65536)  VhT f16 (64x96, cols >=68 zero; row = d, col = key idx)
__global__ void kmattn_attn_kernel(const float* __restrict__ q, const float* __restrict__ k,
                                   const float* __restrict__ v,
                                   const float* __restrict__ memk, const float* __restrict__ memv,
                                   const int* __restrict__ idxq, const int* __restrict__ idxkv,
                                   float* __restrict__ numer, float* __restrict__ denom) {
  __shared__ __align__(16) char smem[65536];
  float*    Kstage = (float*)(smem + 0);
  float*    Vstage = (float*)(smem + 17408);
  float*    Ssm    = (float*)(smem + 0);
  _Float16* Ph     = (_Float16*)(smem + 20480);
  _Float16* Qh     = (_Float16*)(smem + 34816);
  _Float16* Kh     = (_Float16*)(smem + 43008);
  _Float16* VhT    = (_Float16*)(smem + 53248);

  int tid = threadIdx.x;
  int bid = blockIdx.x;
  int c  = bid % NCC;
  int bh = bid / NCC;
  int h  = bh % HH;
  const float* qb  = q + (size_t)bh * TT * DD;
  const float* kb  = k + (size_t)bh * TT * DD;
  const float* vb  = v + (size_t)bh * TT * DD;
  const float* mkb = memk + ((size_t)h * NCC + c) * MMEM * DD;
  const float* mvb = memv + ((size_t)h * NCC + c) * MMEM * DD;
  const size_t ibase = ((size_t)bh * NCC + c) * WW;
  const int* iq = idxq + ibase;
  const int* ik = idxkv + ibase;

  // Q gather: global f32 -> cvt -> Qh f16 (vectorized float4 reads)
  for (int i = tid; i < WW * (DD / 4); i += 256) {
    int r = i >> 4, c4 = i & 15;
    float4 xv = *(const float4*)(qb + (size_t)iq[r] * DD + c4 * 4);
    int o = r * DD + c4 * 4;
    Qh[o + 0] = (_Float16)xv.x; Qh[o + 1] = (_Float16)xv.y;
    Qh[o + 2] = (_Float16)xv.z; Qh[o + 3] = (_Float16)xv.w;
  }

  // K/V gather of the 68 valid rows into f32 staging via async global->LDS
#if HAVE_ASYNC_LDS
  for (int i = tid; i < LK * (DD / 4); i += 256) {
    int r = i >> 4, c4 = i & 15;
    const float* g = (r < MMEM) ? (mkb + r * DD + c4 * 4)
                                : (kb + (size_t)ik[r - MMEM] * DD + c4 * 4);
    __builtin_amdgcn_global_load_async_to_lds_b128((async_b128_t*)g, (async_b128_t*)(Kstage + i * 4), 0, 0);
  }
  for (int i = tid; i < LK * (DD / 4); i += 256) {
    int r = i >> 4, c4 = i & 15;
    const float* g = (r < MMEM) ? (mvb + r * DD + c4 * 4)
                                : (vb + (size_t)ik[r - MMEM] * DD + c4 * 4);
    __builtin_amdgcn_global_load_async_to_lds_b128((async_b128_t*)g, (async_b128_t*)(Vstage + i * 4), 0, 0);
  }
  __builtin_amdgcn_s_wait_asynccnt(0);
#else
  for (int i = tid; i < LK * (DD / 4); i += 256) {
    int r = i >> 4, c4 = i & 15;
    const float* g = (r < MMEM) ? (mkb + r * DD + c4 * 4)
                                : (kb + (size_t)ik[r - MMEM] * DD + c4 * 4);
    *(float4*)(Kstage + i * 4) = *(const float4*)g;
  }
  for (int i = tid; i < LK * (DD / 4); i += 256) {
    int r = i >> 4, c4 = i & 15;
    const float* g = (r < MMEM) ? (mvb + r * DD + c4 * 4)
                                : (vb + (size_t)ik[r - MMEM] * DD + c4 * 4);
    *(float4*)(Vstage + i * 4) = *(const float4*)g;
  }
#endif
  __syncthreads();

  // Convert staging f32 -> f16 WMMA layouts (K row-major padded; V transposed)
  for (int i = tid; i < LKP * DD; i += 256) {
    int r = i / DD, c0 = i % DD;
    Kh[i] = (r < LK) ? (_Float16)Kstage[r * DD + c0] : (_Float16)0.f;
  }
  for (int i = tid; i < DD * LVP; i += 256) {
    int d0 = i / LVP, kk2 = i % LVP;
    VhT[i] = (kk2 < LK) ? (_Float16)Vstage[kk2 * DD + d0] : (_Float16)0.f;
  }
  __syncthreads();

  int wave = tid >> 5, lane = tid & 31;
  // S = Q @ Kh^T : 64 x 80, 4x5 tiles over 8 waves, K = 64 (2 steps of 32)
  for (int tile = wave; tile < 20; tile += 8) {
    int mi = tile / 5, ni = tile % 5;
    CF acc;
#pragma unroll
    for (int r = 0; r < 8; ++r) acc.e[r] = 0.f;
#pragma unroll
    for (int kk = 0; kk < 2; ++kk) {
      v16h a = load_a(Qh, DD, mi * 16, kk * 32, lane);
      v16h b = load_b_nt(Kh, DD, ni * 16, kk * 32, lane);
      acc.v = __builtin_amdgcn_wmma_f32_16x16x32_f16(false, a, false, b, (short)0, acc.v, false, false);
    }
    int mrow = mi * 16 + ((lane >> 4) & 1) * 8;
    int ncol = ni * 16 + (lane & 15);
#pragma unroll
    for (int r = 0; r < 8; ++r) Ssm[(mrow + r) * LKP + ncol] = acc.e[r];
  }
  __syncthreads();

  // Row softmax over 68 valid columns (scale = D^-0.5 = 0.125)
  if (tid < WW) {
    float mx = -FLT_MAX;
    for (int j = 0; j < LK; ++j) mx = fmaxf(mx, Ssm[tid * LKP + j] * 0.125f);
    float sum = 0.f;
    for (int j = 0; j < LK; ++j) {
      float e = __expf(Ssm[tid * LKP + j] * 0.125f - mx);
      Ssm[tid * LKP + j] = e;
      sum += e;
    }
    float rs = 1.0f / sum;
    for (int j = 0; j < LK; ++j)   Ph[tid * LVP + j] = (_Float16)(Ssm[tid * LKP + j] * rs);
    for (int j = LK; j < LVP; ++j) Ph[tid * LVP + j] = (_Float16)0.f;
  }
  __syncthreads();

  // O = P @ V : 64 x 64, 4x4 tiles over 8 waves, K = 96 (3 steps of 32)
  for (int tile = wave; tile < 16; tile += 8) {
    int mi = tile / 4, ni = tile % 4;
    CF acc;
#pragma unroll
    for (int r = 0; r < 8; ++r) acc.e[r] = 0.f;
#pragma unroll
    for (int kk = 0; kk < 3; ++kk) {
      v16h a = load_a(Ph, LVP, mi * 16, kk * 32, lane);
      v16h b = load_b_nt(VhT, LVP, ni * 16, kk * 32, lane);
      acc.v = __builtin_amdgcn_wmma_f32_16x16x32_f16(false, a, false, b, (short)0, acc.v, false, false);
    }
    int mrow = mi * 16 + ((lane >> 4) & 1) * 8;
    int ncol = ni * 16 + (lane & 15);
#pragma unroll
    for (int r = 0; r < 8; ++r) {
      int token = iq[mrow + r];
      atomicAdd(&numer[((size_t)bh * TT + token) * DD + ncol], acc.e[r]);
    }
  }
  if (tid < WW) atomicAdd(&denom[(size_t)bh * TT + iq[tid]], 1.0f);
}

// ---- Kernel D: finalize scatter-mean + aux loss ---------------------------
__global__ void kmattn_final_kernel(const float* __restrict__ numer, const float* __restrict__ denom,
                                    const float* __restrict__ loss, float* __restrict__ out) {
  const size_t N = (size_t)BB * HH * TT * DD;
  size_t i = (size_t)blockIdx.x * 256 + threadIdx.x;
  if (i < N) out[i] = numer[i] / (denom[i / DD] + 1e-5f);
  if (i == 0) out[N] = loss[0] * (0.0001f / (float)((size_t)BB * HH * 2 * TT * DD));
}

extern "C" void kernel_launch(void* const* d_in, const int* in_sizes, int n_in,
                              void* d_out, int out_size, void* d_ws, size_t ws_size,
                              hipStream_t stream) {
  (void)in_sizes; (void)n_in; (void)out_size; (void)ws_size;
  const float* q     = (const float*)d_in[0];
  const float* k     = (const float*)d_in[1];
  const float* v     = (const float*)d_in[2];
  const float* means = (const float*)d_in[3];
  const float* memk  = (const float*)d_in[4];
  const float* memv  = (const float*)d_in[5];
  float* out = (float*)d_out;

  const size_t NUM = (size_t)BB * HH * TT * DD;   // 8,388,608
  const size_t DEN = (size_t)BB * HH * TT;        // 131,072
  const size_t IDX = (size_t)BB * HH * NCC * WW;  // 131,072
  float* numer = (float*)d_ws;
  float* denom = numer + NUM;
  int*   idxq  = (int*)(denom + DEN);
  int*   idxkv = idxq + IDX;
  float* loss  = (float*)(idxkv + IDX);

  (void)hipMemsetAsync(numer, 0, (NUM + DEN) * sizeof(float), stream);
  (void)hipMemsetAsync(loss, 0, sizeof(float), stream);

  kmattn_aux_kernel<<<(BB * HH * 2 * TT) / 256, 256, 0, stream>>>(q, k, means, loss);
  kmattn_select_kernel<<<2 * BB * HH * (NCC / CPB), 256, 0, stream>>>(q, k, means, idxq, idxkv);
  kmattn_attn_kernel<<<BB * HH * NCC, 256, 0, stream>>>(q, k, v, memk, memv, idxq, idxkv, numer, denom);
  kmattn_final_kernel<<<(int)((NUM + 255) / 256), 256, 0, stream>>>(numer, denom, loss, out);
}
